// SimplerAfterShockGNN_44212393345480
// MI455X (gfx1250) — matrix operand order repaired and checked by reading, not compiled, not run on hardware
//
#include <hip/hip_runtime.h>

#define NN 50000
#define NE 800000
#define NG 128
#define MCH 16
#define WCH 128
#define HD 64
#define EPSF 1e-5f

typedef __attribute__((ext_vector_type(16))) _Float16 v16h;
typedef __attribute__((ext_vector_type(8)))  float    v8f;

// ---------------------------------------------------------------------------
// Weight pre-pack: W[K,64] f32 row-major -> per-fragment f16 layout.
// Packed index: (((kt*4 + nt)*32 + lane)*16 + h)
//   lane 0-15  -> col nt*16+lane,    K rows kt*32 + 0..7 (h<8), 16..23 (h>=8)
//   lane 16-31 -> col nt*16+lane-16, K rows kt*32 + 8..15 (h<8), 24..31 (h>=8)
// Zero-pads K up to a multiple of 32.
// ---------------------------------------------------------------------------
__global__ void pack_w_f16(const float* __restrict__ W, _Float16* __restrict__ P, int K) {
    int i = blockIdx.x * blockDim.x + threadIdx.x;
    int KT = (K + 31) / 32;
    int total = KT * 2048;
    if (i >= total) return;
    int idx = i;
    int h    = idx & 15; idx >>= 4;
    int lane = idx & 31; idx >>= 5;
    int nt   = idx & 3;  idx >>= 2;
    int kt   = idx;
    int mlo = lane & 15, hi = lane >> 4;
    int kk = kt * 32 + (h < 8 ? hi * 8 + h : 16 + hi * 8 + (h - 8));
    int c  = nt * 16 + mlo;
    P[i] = (kk < K) ? (_Float16)W[kk * 64 + c] : (_Float16)0.0f;
}

// ---------------------------------------------------------------------------
// WMMA GEMM: out[row, col0+c] = (relu?)(A[row,:K] @ W[:K,64] (+ bias))
// A: [n_rows,K] f32. Wp: packed f16 (see above), staged in LDS via the
// CDNA5 async global->LDS DMA path (ASYNCcnt).
// 256 threads = 8 waves; each wave owns a 16x64 tile. K is compile-time.
// ---------------------------------------------------------------------------
template <int K, bool RELU, bool HASBIAS>
__global__ __launch_bounds__(256)
void gemm_wmma(const float* __restrict__ A, const _Float16* __restrict__ Wp,
               const float* __restrict__ bias, float* __restrict__ out,
               int n_rows, int ldo, int col0)
{
    constexpr int KT = (K + 31) / 32;
    __shared__ alignas(32) _Float16 Wl[KT * 2048];

    const int tid = threadIdx.x;
    {   // async DMA the packed weight panel straight into LDS (no VGPR bounce)
        constexpr int NCH = KT * 2048 / 8;   // 16-byte chunks
        #pragma unroll
        for (int i = 0; i < (NCH + 255) / 256; ++i) {
            int j = i * 256 + tid;
            if (j < NCH) {
                const uint4* src = (const uint4*)Wp + j;
                unsigned ldsoff = (unsigned)(uintptr_t)((uint4*)Wl + j);
                asm volatile("global_load_async_to_lds_b128 %0, %1, off"
                             :: "v"(ldsoff), "v"(src) : "memory");
            }
        }
        asm volatile("s_wait_asynccnt 0" ::: "memory");
    }
    __syncthreads();

    const int wid  = tid >> 5;
    const int lane = tid & 31;
    const long tile = (long)blockIdx.x * 8 + wid;
    const long row0 = tile * 16;
    if (row0 >= n_rows) return;

    const int mlo  = lane & 15;     // A row / out col within tile
    const int hi   = lane >> 4;
    const int koff = hi ? 8 : 0;
    const float* Arow = A + (row0 + mlo) * (long)K;

    v8f acc[4];
    #pragma unroll
    for (int nt = 0; nt < 4; ++nt) {
        const float bv = HASBIAS ? bias[nt * 16 + mlo] : 0.0f;
        #pragma unroll
        for (int r = 0; r < 8; ++r) acc[nt][r] = bv;
    }

    #pragma unroll
    for (int kt = 0; kt < KT; ++kt) {
        const int k0 = kt * 32;
        // A fragment (16-bit A 16x32 layout):
        //   halves 0..7  = K rows k0+koff .. k0+koff+7
        //   halves 8..15 = K rows k0+16+koff .. k0+16+koff+7
        float4 p0 = {0,0,0,0}, p1 = {0,0,0,0}, q0 = {0,0,0,0}, q1 = {0,0,0,0};
        if (k0 < K) {          // compile-time (K mult of 16, koff < 16)
            const float4* ap = (const float4*)(Arow + k0 + koff);
            p0 = ap[0]; p1 = ap[1];
        }
        if (k0 + 16 < K) {     // compile-time
            const float4* aq = (const float4*)(Arow + k0 + 16 + koff);
            q0 = aq[0]; q1 = aq[1];
        }
        v16h a;
        a[0]  = (_Float16)p0.x; a[1]  = (_Float16)p0.y;
        a[2]  = (_Float16)p0.z; a[3]  = (_Float16)p0.w;
        a[4]  = (_Float16)p1.x; a[5]  = (_Float16)p1.y;
        a[6]  = (_Float16)p1.z; a[7]  = (_Float16)p1.w;
        a[8]  = (_Float16)q0.x; a[9]  = (_Float16)q0.y;
        a[10] = (_Float16)q0.z; a[11] = (_Float16)q0.w;
        a[12] = (_Float16)q1.x; a[13] = (_Float16)q1.y;
        a[14] = (_Float16)q1.z; a[15] = (_Float16)q1.w;

        const v16h* bl = (const v16h*)(Wl + kt * 2048);
        #pragma unroll
        for (int nt = 0; nt < 4; ++nt) {
            v16h b = bl[nt * 32 + lane];   // 16 contiguous halves per lane
            acc[nt] = __builtin_amdgcn_wmma_f32_16x16x32_f16(
                false, a, false, b, (short)0, acc[nt], false, false);
        }
    }

    // C/D layout: lanes 0-15: N=lane, VGPR r -> M=r; lanes 16-31: M=8+r
    #pragma unroll
    for (int nt = 0; nt < 4; ++nt) {
        #pragma unroll
        for (int r = 0; r < 8; ++r) {
            const long rr = row0 + (hi ? r + 8 : r);
            float v = acc[nt][r];
            if (RELU) v = fmaxf(v, 0.0f);
            out[rr * (long)ldo + col0 + nt * 16 + mlo] = v;
        }
    }
}

// ---------------------------------------------------------------------------
// Helper kernels
// ---------------------------------------------------------------------------
__global__ void fill_f32(float* __restrict__ p, float v, int n) {
    int i = blockIdx.x * blockDim.x + threadIdx.x;
    if (i < n) p[i] = v;
}

__global__ void deg_edges(const int* __restrict__ ei, float* __restrict__ deg) {
    int e = blockIdx.x * blockDim.x + threadIdx.x;
    if (e < NE) atomicAdd(&deg[ei[NE + e]], 1.0f);   // col = ei[1][e]
}

__global__ void deg_rsqrt(float* __restrict__ d) {
    int i = blockIdx.x * blockDim.x + threadIdx.x;
    if (i < NN) d[i] = rsqrtf(d[i]);                 // deg >= 1 (self loop)
}

// y[n,f] = b[f] + dinv[n]^2 * h[n,f]   (bias + self-loop term)
__global__ void gcn_init(const float* __restrict__ h, const float* __restrict__ dinv,
                         const float* __restrict__ b, float* __restrict__ y) {
    long i = (long)blockIdx.x * blockDim.x + threadIdx.x;
    if (i < (long)NN * HD) {
        int n = (int)(i >> 6), f = (int)(i & 63);
        float di = dinv[n];
        y[i] = b[f] + di * di * h[i];
    }
}

// y[col,f] += dinv[row]*dinv[col] * h[row,f]  over all edges
__global__ __launch_bounds__(256)
void gcn_scatter(const int* __restrict__ ei, const float* __restrict__ dinv,
                 const float* __restrict__ h, float* __restrict__ y) {
    int t = threadIdx.x;
    long e = (long)blockIdx.x * 4 + (t >> 6);
    int f = t & 63;
    if (e < NE) {
        int r = ei[e], c = ei[NE + e];
        float w = dinv[r] * dinv[c];
        atomicAdd(&y[(long)c * HD + f], w * h[(long)r * HD + f]);
    }
}

// per-feature mean / rstd over N nodes; optional residual added first
__global__ __launch_bounds__(256)
void bn_stats(const float* __restrict__ x, const float* __restrict__ res,
              float* __restrict__ mean, float* __restrict__ rstd) {
    __shared__ float s1[256], s2[256];
    int f = blockIdx.x, t = threadIdx.x;
    float a = 0.0f, b = 0.0f;
    for (int n = t; n < NN; n += 256) {
        float v = x[(long)n * HD + f];
        if (res) v += res[(long)n * HD + f];
        a += v; b += v * v;
    }
    s1[t] = a; s2[t] = b;
    __syncthreads();
    for (int s = 128; s > 0; s >>= 1) {
        if (t < s) { s1[t] += s1[t + s]; s2[t] += s2[t + s]; }
        __syncthreads();
    }
    if (t == 0) {
        float mu = s1[0] / (float)NN;
        float var = s2[0] / (float)NN - mu * mu;
        mean[f] = mu;
        rstd[f] = rsqrtf(var + EPSF);
    }
}

__global__ void bn_apply_relu(const float* __restrict__ x, const float* __restrict__ res,
                              const float* __restrict__ mean, const float* __restrict__ rstd,
                              const float* __restrict__ g, const float* __restrict__ be,
                              float* __restrict__ out) {
    long i = (long)blockIdx.x * blockDim.x + threadIdx.x;
    if (i < (long)NN * HD) {
        int f = (int)(i & 63);
        float v = x[i];
        if (res) v += res[i];
        v = (v - mean[f]) * rstd[f] * g[f] + be[f];
        out[i] = fmaxf(v, 0.0f);
    }
}

__global__ void pool_counts(const int* __restrict__ batch, float* __restrict__ counts) {
    int n = blockIdx.x * blockDim.x + threadIdx.x;
    if (n < NN) atomicAdd(&counts[batch[n]], 1.0f);
}

// xc[g, 0:64] += x2[n];  xc[g, 64:128] = max(xc, x2[n])   (x2 >= 0 post-ReLU)
__global__ void pool_scatter(const int* __restrict__ batch, const float* __restrict__ x2,
                             float* __restrict__ xc) {
    long i = (long)blockIdx.x * blockDim.x + threadIdx.x;
    if (i < (long)NN * HD) {
        int n = (int)(i >> 6), f = (int)(i & 63);
        int g = batch[n];
        float v = x2[i];
        atomicAdd(&xc[(long)g * 128 + f], v);
        atomicMax((unsigned int*)&xc[(long)g * 128 + 64 + f], __float_as_uint(v));
    }
}

__global__ void pool_finish(const float* __restrict__ counts, float* __restrict__ xc) {
    int i = blockIdx.x * blockDim.x + threadIdx.x;   // G*64 threads
    if (i < NG * HD) {
        int g = i >> 6, f = i & 63;
        xc[(long)g * 128 + f] /= fmaxf(counts[g], 1.0f);
    }
}

// final dot: out[head*G + g] = sum_t hid[g,t] * W2[t] + b2
__global__ __launch_bounds__(64)
void head_out(const float* __restrict__ hidA, const float* __restrict__ hidB,
              const float* __restrict__ W2a, const float* __restrict__ b2a,
              const float* __restrict__ W2b, const float* __restrict__ b2b,
              float* __restrict__ out) {
    __shared__ float red[64];
    int g = blockIdx.x % NG;
    int head = blockIdx.x / NG;
    const float* hid = head ? hidB : hidA;
    const float* W2  = head ? W2b : W2a;
    const float* b2  = head ? b2b : b2a;
    int t = threadIdx.x;
    red[t] = hid[(long)g * HD + t] * W2[t];
    __syncthreads();
    for (int s = 32; s > 0; s >>= 1) {
        if (t < s) red[t] += red[t + s];
        __syncthreads();
    }
    if (t == 0) out[head * NG + g] = red[0] + b2[0];
}

// ---------------------------------------------------------------------------
extern "C" void kernel_launch(void* const* d_in, const int* in_sizes, int n_in,
                              void* d_out, int out_size, void* d_ws, size_t ws_size,
                              hipStream_t stream) {
    const float* metadata = (const float*)d_in[0];
    const float* wavef    = (const float*)d_in[1];
    const int*   ei       = (const int*)d_in[2];   // [2,E] int32
    const int*   batch    = (const int*)d_in[3];
    const float* Wm  = (const float*)d_in[4];  const float* bm  = (const float*)d_in[5];
    const float* Ww  = (const float*)d_in[6];  const float* bw  = (const float*)d_in[7];
    const float* Wc  = (const float*)d_in[8];  const float* bc  = (const float*)d_in[9];
    const float* W1  = (const float*)d_in[10]; const float* b1  = (const float*)d_in[11];
    const float* W2  = (const float*)d_in[12]; const float* b2  = (const float*)d_in[13];
    const float* g1  = (const float*)d_in[14]; const float* be1 = (const float*)d_in[15];
    const float* g2  = (const float*)d_in[16]; const float* be2 = (const float*)d_in[17];
    const float* Wla1 = (const float*)d_in[18]; const float* bla1 = (const float*)d_in[19];
    const float* Wla2 = (const float*)d_in[20]; const float* bla2 = (const float*)d_in[21];
    const float* Wlo1 = (const float*)d_in[22]; const float* blo1 = (const float*)d_in[23];
    const float* Wlo2 = (const float*)d_in[24]; const float* blo2 = (const float*)d_in[25];
    float* out = (float*)d_out;

    // --- workspace layout (floats) ---
    const long N64 = (long)NN * HD;
    float* ws   = (float*)d_ws;
    float* dinv = ws;                       // N
    float* cat  = dinv + NN;                // N*128  (me || we)
    float* B0   = cat + (long)NN * 128;     // N*64   x      -> later y2
    float* B1   = B0 + N64;                 // N*64   h      -> later h2
    float* B2   = B1 + N64;                 // N*64   y      -> later x2
    float* B3   = B2 + N64;                 // N*64   x1
    float* xc     = B3 + N64;               // G*128 pooled (mean||max)
    float* counts = xc + (long)NG * 128;    // G
    float* mean1 = counts + NG;             // 64
    float* rstd1 = mean1 + 64;              // 64
    float* mean2 = rstd1 + 64;              // 64
    float* rstd2 = mean2 + 64;              // 64
    float* hidLat = rstd2 + 64;             // G*64
    float* hidLon = hidLat + (long)NG * HD; // G*64
    // packed f16 weight panels (sizes: ceil(K/32)*2048 halves), 16B aligned
    _Float16* f16base = (_Float16*)(hidLon + (long)NG * HD);
    _Float16* Wm16 = f16base;               // KT=1 -> 2048
    _Float16* Ww16 = Wm16 + 2048;           // KT=4 -> 8192
    _Float16* Wc16 = Ww16 + 8192;           // 8192
    _Float16* W116 = Wc16 + 8192;           // KT=2 -> 4096
    _Float16* W216 = W116 + 4096;           // 4096
    _Float16* Wla16 = W216 + 4096;          // 8192
    _Float16* Wlo16 = Wla16 + 8192;         // 8192

    // --- weight packing f32 -> fragment-ordered f16 ---
    pack_w_f16<<<(2048 + 255) / 256, 256, 0, stream>>>(Wm, Wm16, MCH);
    pack_w_f16<<<(8192 + 255) / 256, 256, 0, stream>>>(Ww, Ww16, WCH);
    pack_w_f16<<<(8192 + 255) / 256, 256, 0, stream>>>(Wc, Wc16, 128);
    pack_w_f16<<<(4096 + 255) / 256, 256, 0, stream>>>(W1, W116, 64);
    pack_w_f16<<<(4096 + 255) / 256, 256, 0, stream>>>(W2, W216, 64);
    pack_w_f16<<<(8192 + 255) / 256, 256, 0, stream>>>(Wla1, Wla16, 128);
    pack_w_f16<<<(8192 + 255) / 256, 256, 0, stream>>>(Wlo1, Wlo16, 128);

    // --- degree / symmetric normalization ---
    fill_f32<<<(NN + 255) / 256, 256, 0, stream>>>(dinv, 1.0f, NN);   // self loop
    deg_edges<<<(NE + 255) / 256, 256, 0, stream>>>(ei, dinv);
    deg_rsqrt<<<(NN + 255) / 256, 256, 0, stream>>>(dinv);

    const int gemmBlocksN = (NN / 16 + 7) / 8;       // 3125 tiles / 8 waves
    const int ewBlocks = (int)((N64 + 255) / 256);

    // --- encoders ---
    gemm_wmma<16,  true, true><<<gemmBlocksN, 256, 0, stream>>>(metadata, Wm16, bm, cat, NN, 128, 0);
    gemm_wmma<128, true, true><<<gemmBlocksN, 256, 0, stream>>>(wavef,    Ww16, bw, cat, NN, 128, 64);
    gemm_wmma<128, true, true><<<gemmBlocksN, 256, 0, stream>>>(cat,      Wc16, bc, B0,  NN, 64,  0);   // x = B0

    // --- conv 1 ---
    gemm_wmma<64, false, false><<<gemmBlocksN, 256, 0, stream>>>(B0, W116, nullptr, B1, NN, 64, 0);     // h = B1
    gcn_init<<<ewBlocks, 256, 0, stream>>>(B1, dinv, b1, B2);                                           // y = B2
    gcn_scatter<<<(NE + 3) / 4, 256, 0, stream>>>(ei, dinv, B1, B2);
    bn_stats<<<64, 256, 0, stream>>>(B2, nullptr, mean1, rstd1);
    bn_apply_relu<<<ewBlocks, 256, 0, stream>>>(B2, nullptr, mean1, rstd1, g1, be1, B3);                // x1 = B3

    // --- conv 2 (+ residual) ---
    gemm_wmma<64, false, false><<<gemmBlocksN, 256, 0, stream>>>(B3, W216, nullptr, B1, NN, 64, 0);     // h2 = B1
    gcn_init<<<ewBlocks, 256, 0, stream>>>(B1, dinv, b2, B0);                                           // y2 = B0
    gcn_scatter<<<(NE + 3) / 4, 256, 0, stream>>>(ei, dinv, B1, B0);
    bn_stats<<<64, 256, 0, stream>>>(B0, B3, mean2, rstd2);
    bn_apply_relu<<<ewBlocks, 256, 0, stream>>>(B0, B3, mean2, rstd2, g2, be2, B2);                     // x2 = B2

    // --- pooling ---
    fill_f32<<<(NG * 128 + 255) / 256, 256, 0, stream>>>(xc, 0.0f, NG * 128);
    fill_f32<<<(NG + 255) / 256, 256, 0, stream>>>(counts, 0.0f, NG);
    pool_counts<<<(NN + 255) / 256, 256, 0, stream>>>(batch, counts);
    pool_scatter<<<ewBlocks, 256, 0, stream>>>(batch, B2, xc);
    pool_finish<<<(NG * HD + 255) / 256, 256, 0, stream>>>(counts, xc);

    // --- heads (first layer via WMMA, K=128) ---
    const int gemmBlocksG = (NG / 16 + 7) / 8;   // = 1
    gemm_wmma<128, true, true><<<gemmBlocksG, 256, 0, stream>>>(xc, Wla16, bla1, hidLat, NG, 64, 0);
    gemm_wmma<128, true, true><<<gemmBlocksG, 256, 0, stream>>>(xc, Wlo16, blo1, hidLon, NG, 64, 0);
    head_out<<<2 * NG, 64, 0, stream>>>(hidLat, hidLon, Wla2, bla2, Wlo2, blo2, out);
}
